// WordSAGE_52922587021522
// MI455X (gfx1250) — compile-verified, matrix operand order checked by reading, not run on hardware
//
#include <hip/hip_runtime.h>
#include <hip/hip_bf16.h>

// ---------------------------------------------------------------------------
// WordSAGE pipeline for MI455X (gfx1250), wave32 + WMMA bf16.
//   N=4096 nodes, C=2500 feats, E=131072 edges. GEMMs (~630 GFLOP) dominate;
//   operands fit in 192MB L2, so we run bf16 WMMA with 32x64 tiles per wave
//   (8 v_wmma per K-step, branch-free inner loop) for matrix-core-bound exec.
// ---------------------------------------------------------------------------

#define NN   4096
#define EE   131072
#define CC   2500
#define C3   7500
#define KP   2528      // CC rounded up to multiple of 32 (79*32)
#define KATT 4096      // K for attn@V gemm (N, already mult of 32)
#define NBCE 160
#define NCE  16

typedef __attribute__((ext_vector_type(16))) __bf16 v16bfx;
typedef __attribute__((ext_vector_type(8)))  float  v8fx;

struct alignas(16) U128 { unsigned int w[4]; };
union FragBF { U128 q[2]; v16bfx v; };

__device__ __forceinline__ float leakyf(float x) { return x >= 0.f ? x : 0.01f * x; }

// ------------------------------- CSR build ---------------------------------
__global__ void k_zero_i32(int* p, int n) {
    int i = blockIdx.x * blockDim.x + threadIdx.x;
    if (i < n) p[i] = 0;
}
__global__ void k_count_edges(const int* __restrict__ ei, int* __restrict__ cnt) {
    int e = blockIdx.x * blockDim.x + threadIdx.x;
    if (e < EE) atomicAdd(&cnt[ei[EE + e]], 1);
}
__global__ void k_scan_serial(const int* __restrict__ cnt, int* __restrict__ offs) {
    if (threadIdx.x == 0 && blockIdx.x == 0) {
        int acc = 0;
        for (int i = 0; i < NN; ++i) { offs[i] = acc; acc += cnt[i]; }
        offs[NN] = acc;
    }
}
__global__ void k_fill_adj(const int* __restrict__ ei, const int* __restrict__ offs,
                           int* __restrict__ cursor, int* __restrict__ adj) {
    int e = blockIdx.x * blockDim.x + threadIdx.x;
    if (e < EE) {
        int d = ei[EE + e];
        int p = atomicAdd(&cursor[d], 1);
        adj[offs[d] + p] = ei[e];
    }
}

// ----------------------- neighbor mean (gather; x in L2) -------------------
// Adjacency list is staged into LDS with the gfx1250 async global->LDS path
// (ASYNCcnt-tracked); the C-wide gather loop then reads indices from LDS.
#define ADJ_STAGE 1024
__global__ void __launch_bounds__(256)
k_agg_mean(const int* __restrict__ offs, const int* __restrict__ adj,
           const float* __restrict__ X, int ld, float* __restrict__ out) {
    __shared__ int s_adj[ADJ_STAGE];
    int node = blockIdx.x;
    int beg = offs[node], end = offs[node + 1];
    int deg = end - beg;
    int degS = deg < ADJ_STAGE ? deg : ADJ_STAGE;

    for (int e = threadIdx.x; e < degS; e += blockDim.x) {
        unsigned lofs = (unsigned)(size_t)(&s_adj[e]);           // LDS byte offset (low 32b)
        unsigned long long ga = (unsigned long long)(adj + beg + e);
        asm volatile("global_load_async_to_lds_b32 %0, %1, off"
                     :: "v"(lofs), "v"(ga) : "memory");
    }
    asm volatile("s_wait_asynccnt 0x0" ::: "memory");
    __syncthreads();

    float inv = (deg > 0) ? 1.f / (float)deg : 0.f;   // deg==0 -> 0 (matches s/max(cnt,1))
    for (int c = threadIdx.x; c < CC; c += blockDim.x) {
        float acc = 0.f;
        for (int e = 0; e < degS; ++e)
            acc += X[(long)s_adj[e] * ld + c];
        for (int e = beg + degS; e < end; ++e)        // overflow fallback (rare)
            acc += X[(long)adj[e] * ld + c];
        out[(long)node * CC + c] = acc * inv;
    }
}

// --------------------------- f32 -> bf16 (padded) --------------------------
__global__ void k_cast_bf16(const float* __restrict__ src, int srcLd, int colOff,
                            __bf16* __restrict__ dst, int rows, int cols, int dstLd) {
    long i = (long)blockIdx.x * blockDim.x + threadIdx.x;
    long total = (long)rows * dstLd;
    if (i >= total) return;
    int r = (int)(i / dstLd), c = (int)(i % dstLd);
    float v = (c < cols) ? src[(long)r * srcLd + colOff + c] : 0.f;
    dst[i] = (__bf16)v;
}
// dst[c * dstLd + r] = src[r * srcLd + colOff + c]   (V -> V^T for attn@V)
__global__ void k_transpose_cast_bf16(const float* __restrict__ src, int srcLd, int colOff,
                                      __bf16* __restrict__ dst, int rows, int cols, int dstLd) {
    long i = (long)blockIdx.x * blockDim.x + threadIdx.x;
    long total = (long)rows * cols;
    if (i >= total) return;
    int c = (int)(i / rows), r = (int)(i % rows);
    dst[(long)c * dstLd + r] = (__bf16)src[(long)r * srcLd + colOff + c];
}

// ------------------------- WMMA bf16 GEMM: C = A*B^T -----------------------
// A: [M, K] bf16 (lda), B: [Nout, K] bf16 (ldb), Cout: [M, ldc] f32.
// flags: 1 = accumulate onto existing Cout, 2 = add bias[col] on init.
// 32x64 tile per wave: 2 A-frags x 4 B-frags -> 8 v_wmma per 32-K step.
// Edge columns are CLAMPED on load (garbage accumulated, masked at store)
// so the K loop is branch-free and EXEC stays all-ones (WMMA requirement).
// Requires M % 32 == 0, K % 32 == 0.
__global__ void __launch_bounds__(256)
k_gemm_bf16_wmma(const __bf16* __restrict__ A, int lda,
                 const __bf16* __restrict__ B, int ldb,
                 const float* __restrict__ bias,
                 float* __restrict__ Cout, int ldc,
                 int M, int Nout, int K, int flags) {
    const int lane = threadIdx.x & 31;
    const int wave = threadIdx.x >> 5;
    const int tilesN = (Nout + 63) >> 6;
    const int tilesM = M >> 5;
    int t = blockIdx.x * 8 + wave;
    if (t >= tilesM * tilesN) return;                 // wave-uniform
    const int tm = t / tilesN, tn = t % tilesN;
    const int row0 = tm << 5, col0 = tn << 6;
    const int hlf = lane >> 4;                        // 0 | 1 half-wave
    const int l15 = lane & 15;

    // 16-bit A fragment layout (ISA 7.12.2): lanes 0-15 hold K {0..7,16..23},
    // lanes 16-31 hold K {8..15,24..31} -> two 16B chunks at +0/+16 elements.
    const __bf16* aB0 = A + (long)(row0 + l15) * lda + hlf * 8;       // rows row0..+15
    const __bf16* aB1 = aB0 + (long)16 * lda;                         // rows row0+16..+31

    int bcol[4];
    const __bf16* bB[4];
    #pragma unroll
    for (int j = 0; j < 4; ++j) {
        int c = col0 + j * 16 + l15;
        bcol[j] = c;
        int cc = (c < Nout) ? c : (Nout - 1);         // clamp: branch-free loads
        bB[j] = B + (long)cc * ldb + hlf * 8;
    }

    v8fx acc[2][4];
    #pragma unroll
    for (int i = 0; i < 2; ++i)
        #pragma unroll
        for (int j = 0; j < 4; ++j) {
            if (flags & 1) {
                #pragma unroll
                for (int k = 0; k < 8; ++k) {
                    int r = row0 + i * 16 + (hlf << 3) + k;
                    acc[i][j][k] = (bcol[j] < Nout) ? Cout[(long)r * ldc + bcol[j]] : 0.f;
                }
            } else {
                float b0 = ((flags & 2) && bcol[j] < Nout) ? bias[bcol[j]] : 0.f;
                #pragma unroll
                for (int k = 0; k < 8; ++k) acc[i][j][k] = b0;
            }
        }

    for (int k0 = 0; k0 < K; k0 += 32) {
        FragBF a0, a1, b0, b1, b2, b3;
        a0.q[0] = *(const U128*)(aB0 + k0); a0.q[1] = *(const U128*)(aB0 + k0 + 16);
        a1.q[0] = *(const U128*)(aB1 + k0); a1.q[1] = *(const U128*)(aB1 + k0 + 16);
        b0.q[0] = *(const U128*)(bB[0] + k0); b0.q[1] = *(const U128*)(bB[0] + k0 + 16);
        b1.q[0] = *(const U128*)(bB[1] + k0); b1.q[1] = *(const U128*)(bB[1] + k0 + 16);
        b2.q[0] = *(const U128*)(bB[2] + k0); b2.q[1] = *(const U128*)(bB[2] + k0 + 16);
        b3.q[0] = *(const U128*)(bB[3] + k0); b3.q[1] = *(const U128*)(bB[3] + k0 + 16);
        // speculative next-K prefetch (SYS scope; dropped on fault per ISA)
        __builtin_prefetch(aB0 + k0 + 32, 0, 0);
        __builtin_prefetch(bB[0] + k0 + 32, 0, 0);
        __builtin_prefetch(bB[2] + k0 + 32, 0, 0);

        acc[0][0] = __builtin_amdgcn_wmma_f32_16x16x32_bf16(false, a0.v, false, b0.v, (short)0, acc[0][0], false, false);
        acc[0][1] = __builtin_amdgcn_wmma_f32_16x16x32_bf16(false, a0.v, false, b1.v, (short)0, acc[0][1], false, false);
        acc[0][2] = __builtin_amdgcn_wmma_f32_16x16x32_bf16(false, a0.v, false, b2.v, (short)0, acc[0][2], false, false);
        acc[0][3] = __builtin_amdgcn_wmma_f32_16x16x32_bf16(false, a0.v, false, b3.v, (short)0, acc[0][3], false, false);
        acc[1][0] = __builtin_amdgcn_wmma_f32_16x16x32_bf16(false, a1.v, false, b0.v, (short)0, acc[1][0], false, false);
        acc[1][1] = __builtin_amdgcn_wmma_f32_16x16x32_bf16(false, a1.v, false, b1.v, (short)0, acc[1][1], false, false);
        acc[1][2] = __builtin_amdgcn_wmma_f32_16x16x32_bf16(false, a1.v, false, b2.v, (short)0, acc[1][2], false, false);
        acc[1][3] = __builtin_amdgcn_wmma_f32_16x16x32_bf16(false, a1.v, false, b3.v, (short)0, acc[1][3], false, false);
    }

    #pragma unroll
    for (int i = 0; i < 2; ++i)
        #pragma unroll
        for (int j = 0; j < 4; ++j)
            if (bcol[j] < Nout) {
                #pragma unroll
                for (int k = 0; k < 8; ++k) {
                    int r = row0 + i * 16 + (hlf << 3) + k;
                    Cout[(long)r * ldc + bcol[j]] = acc[i][j][k];
                }
            }
}

// ------------------------------- epilogues ---------------------------------
__global__ void k_bn_elem(float* __restrict__ h, const float* __restrict__ g,
                          const float* __restrict__ b, const float* __restrict__ m,
                          const float* __restrict__ v, int doLeaky) {
    long i = (long)blockIdx.x * blockDim.x + threadIdx.x;
    if (i >= (long)NN * CC) return;
    int c = (int)(i % CC);
    float x = (h[i] - m[c]) * rsqrtf(v[c] + 1e-5f) * g[c] + b[c];
    h[i] = doLeaky ? leakyf(x) : x;
}
__global__ void __launch_bounds__(256)
k_ln_leaky(float* __restrict__ h, const float* __restrict__ g, const float* __restrict__ b) {
    int row = blockIdx.x;
    float* x = h + (long)row * CC;
    __shared__ float red[8];
    float s = 0.f, s2 = 0.f;
    for (int c = threadIdx.x; c < CC; c += blockDim.x) { float v = x[c]; s += v; s2 += v * v; }
    for (int o = 16; o; o >>= 1) { s += __shfl_xor(s, o, 32); s2 += __shfl_xor(s2, o, 32); }
    if ((threadIdx.x & 31) == 0) red[threadIdx.x >> 5] = s;
    __syncthreads();
    if (threadIdx.x < 8) s = red[threadIdx.x];
    for (int o = 4; o; o >>= 1) s += __shfl_xor(s, o, 32);
    s = __shfl(s, 0, 32);
    __syncthreads();
    if ((threadIdx.x & 31) == 0) red[threadIdx.x >> 5] = s2;
    __syncthreads();
    if (threadIdx.x < 8) s2 = red[threadIdx.x];
    for (int o = 4; o; o >>= 1) s2 += __shfl_xor(s2, o, 32);
    s2 = __shfl(s2, 0, 32);
    float mu = s / (float)CC;
    float var = s2 / (float)CC - mu * mu;
    float inv = rsqrtf(var + 1e-5f);
    for (int c = threadIdx.x; c < CC; c += blockDim.x)
        x[c] = leakyf((x[c] - mu) * inv * g[c] + b[c]);
}
// scores (f32, in-place exp) -> normalized bf16 attn
__global__ void __launch_bounds__(256)
k_softmax_bf16(float* __restrict__ S, __bf16* __restrict__ Abf, float scale) {
    int row = blockIdx.x;
    float* s = S + (long)row * NN;
    __shared__ float red[8];
    float mx = -3.4e38f;
    for (int c = threadIdx.x; c < NN; c += blockDim.x) mx = fmaxf(mx, s[c]);
    for (int o = 16; o; o >>= 1) mx = fmaxf(mx, __shfl_xor(mx, o, 32));
    if ((threadIdx.x & 31) == 0) red[threadIdx.x >> 5] = mx;
    __syncthreads();
    if (threadIdx.x < 8) mx = red[threadIdx.x];
    for (int o = 4; o; o >>= 1) mx = fmaxf(mx, __shfl_xor(mx, o, 32));
    mx = __shfl(mx, 0, 32) * scale;
    float sum = 0.f;
    for (int c = threadIdx.x; c < NN; c += blockDim.x) {
        float e = __expf(s[c] * scale - mx);
        s[c] = e; sum += e;
    }
    for (int o = 16; o; o >>= 1) sum += __shfl_xor(sum, o, 32);
    __syncthreads();
    if ((threadIdx.x & 31) == 0) red[threadIdx.x >> 5] = sum;
    __syncthreads();
    if (threadIdx.x < 8) sum = red[threadIdx.x];
    for (int o = 4; o; o >>= 1) sum += __shfl_xor(sum, o, 32);
    float inv = 1.f / __shfl(sum, 0, 32);
    for (int c = threadIdx.x; c < NN; c += blockDim.x)
        Abf[(long)row * NN + c] = (__bf16)(s[c] * inv);
}
__global__ void k_relu_inplace(float* __restrict__ h, long n) {
    long i = (long)blockIdx.x * blockDim.x + threadIdx.x;
    if (i < n) h[i] = fmaxf(h[i], 0.f);
}
__global__ void k_leaky_copy(const float* __restrict__ src, float* __restrict__ dst, long n) {
    long i = (long)blockIdx.x * blockDim.x + threadIdx.x;
    if (i < n) dst[i] = leakyf(src[i]);
}

// ------------------------------- launcher ----------------------------------
static inline char* wsalloc(char*& p, size_t bytes) {
    char* r = p;
    p += (bytes + 255) & ~(size_t)255;
    return r;
}
static inline int cdiv(long a, long b) { return (int)((a + b - 1) / b); }

static void run_gemm(const __bf16* A, int lda, const __bf16* B, int ldb,
                     const float* bias, float* C, int ldc,
                     int M, int Nout, int K, int flags, hipStream_t st) {
    int tiles = (M >> 5) * ((Nout + 63) >> 6);
    k_gemm_bf16_wmma<<<cdiv(tiles, 8), 256, 0, st>>>(A, lda, B, ldb, bias, C, ldc, M, Nout, K, flags);
}
static void run_cast(const float* s, int sld, int coff, __bf16* d, int rows, int cols, int dld, hipStream_t st) {
    long tot = (long)rows * dld;
    k_cast_bf16<<<cdiv(tot, 256), 256, 0, st>>>(s, sld, coff, d, rows, cols, dld);
}

extern "C" void kernel_launch(void* const* d_in, const int* in_sizes, int n_in,
                              void* d_out, int out_size, void* d_ws, size_t ws_size,
                              hipStream_t stream) {
    const float* x        = (const float*)d_in[0];
    const int*   ei       = (const int*)  d_in[1];
    const float* W_self1  = (const float*)d_in[2];
    const float* W_neigh1 = (const float*)d_in[3];
    const float* b_conv1  = (const float*)d_in[4];
    const float* bn1_g = (const float*)d_in[5],  *bn1_b = (const float*)d_in[6];
    const float* bn1_m = (const float*)d_in[7],  *bn1_v = (const float*)d_in[8];
    const float* Wqkv  = (const float*)d_in[9],  *bqkv  = (const float*)d_in[10];
    const float* Wo    = (const float*)d_in[11], *bo    = (const float*)d_in[12];
    const float* ln_g  = (const float*)d_in[13], *ln_b  = (const float*)d_in[14];
    const float* W_self2  = (const float*)d_in[15];
    const float* W_neigh2 = (const float*)d_in[16];
    const float* b_conv2  = (const float*)d_in[17];
    const float* bn2_g = (const float*)d_in[18], *bn2_b = (const float*)d_in[19];
    const float* bn2_m = (const float*)d_in[20], *bn2_v = (const float*)d_in[21];
    const float* W_lin = (const float*)d_in[22], *b_lin = (const float*)d_in[23];
    const float* W_bce = (const float*)d_in[24], *b_bce = (const float*)d_in[25];
    const float* W_ce  = (const float*)d_in[26], *b_ce  = (const float*)d_in[27];

    float* out_x = (float*)d_out;                 // [N, C]
    float* out_b = out_x + (long)NN * CC;         // [N, 160]
    float* out_y = out_b + (long)NN * NBCE;       // [N, 16]

    // ---- workspace carve-up ----
    char* p = (char*)d_ws;
    int* counts = (int*)wsalloc(p, NN * 4);
    int* offs   = (int*)wsalloc(p, (NN + 1) * 4);
    int* cursor = (int*)wsalloc(p, NN * 4);
    int* adj    = (int*)wsalloc(p, EE * 4);
    __bf16* xb    = (__bf16*)wsalloc(p, (size_t)NN * KP * 2);
    __bf16* Ws1b  = (__bf16*)wsalloc(p, (size_t)CC * KP * 2);
    __bf16* Wn1b  = (__bf16*)wsalloc(p, (size_t)CC * KP * 2);
    __bf16* Wqkvb = (__bf16*)wsalloc(p, (size_t)C3 * KP * 2);
    __bf16* Wob   = (__bf16*)wsalloc(p, (size_t)CC * KP * 2);
    __bf16* Ws2b  = (__bf16*)wsalloc(p, (size_t)CC * KP * 2);
    __bf16* Wn2b  = (__bf16*)wsalloc(p, (size_t)CC * KP * 2);
    __bf16* Wlinb = (__bf16*)wsalloc(p, (size_t)CC * KP * 2);
    __bf16* Wbceb = (__bf16*)wsalloc(p, (size_t)NBCE * KP * 2);
    __bf16* Wceb  = (__bf16*)wsalloc(p, (size_t)NCE * KP * 2);
    float*  agg1  = (float*) wsalloc(p, (size_t)NN * CC * 4);
    __bf16* agg1b = (__bf16*)wsalloc(p, (size_t)NN * KP * 2);
    float*  h1    = (float*) wsalloc(p, (size_t)NN * CC * 4);
    __bf16* h1b   = (__bf16*)wsalloc(p, (size_t)NN * KP * 2);
    float*  qkv   = (float*) wsalloc(p, (size_t)NN * C3 * 4);
    __bf16* qb    = (__bf16*)wsalloc(p, (size_t)NN * KP * 2);
    __bf16* kb    = (__bf16*)wsalloc(p, (size_t)NN * KP * 2);
    __bf16* vT    = (__bf16*)wsalloc(p, (size_t)CC * KATT * 2);
    float*  scores= (float*) wsalloc(p, (size_t)NN * NN * 4);
    __bf16* attnb = (__bf16*)wsalloc(p, (size_t)NN * NN * 2);
    float*  av    = (float*) wsalloc(p, (size_t)NN * CC * 4);
    __bf16* avb   = (__bf16*)wsalloc(p, (size_t)NN * KP * 2);
    float*  hB    = (float*) wsalloc(p, (size_t)NN * CC * 4);
    __bf16* hBb   = (__bf16*)wsalloc(p, (size_t)NN * KP * 2);
    float*  agg2  = (float*) wsalloc(p, (size_t)NN * CC * 4);
    __bf16* agg2b = (__bf16*)wsalloc(p, (size_t)NN * KP * 2);
    float*  h2    = (float*) wsalloc(p, (size_t)NN * CC * 4);
    __bf16* h2b   = (__bf16*)wsalloc(p, (size_t)NN * KP * 2);
    float*  h3    = (float*) wsalloc(p, (size_t)NN * CC * 4);
    __bf16* h3b   = (__bf16*)wsalloc(p, (size_t)NN * KP * 2);
    (void)ws_size; (void)n_in; (void)in_sizes; (void)out_size;

    const long nc = (long)NN * CC;

    // ---- CSR build (gather aggregation keeps x resident in 192MB L2) ----
    k_zero_i32<<<cdiv(NN, 256), 256, 0, stream>>>(counts, NN);
    k_zero_i32<<<cdiv(NN, 256), 256, 0, stream>>>(cursor, NN);
    k_count_edges<<<cdiv(EE, 256), 256, 0, stream>>>(ei, counts);
    k_scan_serial<<<1, 32, 0, stream>>>(counts, offs);
    k_fill_adj<<<cdiv(EE, 256), 256, 0, stream>>>(ei, offs, cursor, adj);

    // ---- bf16 conversions (weights + x) ----
    run_cast(x, CC, 0, xb, NN, CC, KP, stream);
    run_cast(W_self1,  CC, 0, Ws1b,  CC, CC, KP, stream);
    run_cast(W_neigh1, CC, 0, Wn1b,  CC, CC, KP, stream);
    run_cast(Wqkv,     CC, 0, Wqkvb, C3, CC, KP, stream);
    run_cast(Wo,       CC, 0, Wob,   CC, CC, KP, stream);
    run_cast(W_self2,  CC, 0, Ws2b,  CC, CC, KP, stream);
    run_cast(W_neigh2, CC, 0, Wn2b,  CC, CC, KP, stream);
    run_cast(W_lin,    CC, 0, Wlinb, CC, CC, KP, stream);
    run_cast(W_bce,    CC, 0, Wbceb, NBCE, CC, KP, stream);
    run_cast(W_ce,     CC, 0, Wceb,  NCE,  CC, KP, stream);

    // ---- SAGE conv1: h1 = x@Ws1^T + b + mean(x_nbr)@Wn1^T ----
    k_agg_mean<<<NN, 256, 0, stream>>>(offs, adj, x, CC, agg1);
    run_cast(agg1, CC, 0, agg1b, NN, CC, KP, stream);
    run_gemm(xb,    KP, Ws1b, KP, b_conv1, h1, CC, NN, CC, KP, /*BIAS*/2, stream);
    run_gemm(agg1b, KP, Wn1b, KP, nullptr, h1, CC, NN, CC, KP, /*ACC*/1, stream);
    k_bn_elem<<<cdiv(nc, 256), 256, 0, stream>>>(h1, bn1_g, bn1_b, bn1_m, bn1_v, 1);
    run_cast(h1, CC, 0, h1b, NN, CC, KP, stream);

    // ---- attention: qkv -> softmax(q k^T / sqrt(C)) v -> Wo ----
    run_gemm(h1b, KP, Wqkvb, KP, bqkv, qkv, C3, NN, C3, KP, 2, stream);
    run_cast(qkv, C3, 0,    qb, NN, CC, KP, stream);
    run_cast(qkv, C3, CC,   kb, NN, CC, KP, stream);
    k_transpose_cast_bf16<<<cdiv(nc, 256), 256, 0, stream>>>(qkv, C3, 2 * CC, vT, NN, CC, KATT);
    run_gemm(qb, KP, kb, KP, nullptr, scores, NN, NN, NN, KP, 0, stream);
    k_softmax_bf16<<<NN, 256, 0, stream>>>(scores, attnb, 0.02f /* 1/sqrt(2500) */);
    run_gemm(attnb, KATT, vT, KATT, nullptr, av, CC, NN, CC, KATT, 0, stream);
    run_cast(av, CC, 0, avb, NN, CC, KP, stream);
    run_gemm(avb, KP, Wob, KP, bo, hB, CC, NN, CC, KP, 2, stream);
    k_ln_leaky<<<NN, 256, 0, stream>>>(hB, ln_g, ln_b);
    run_cast(hB, CC, 0, hBb, NN, CC, KP, stream);

    // ---- SAGE conv2 + BN2 ----
    k_agg_mean<<<NN, 256, 0, stream>>>(offs, adj, hB, CC, agg2);
    run_cast(agg2, CC, 0, agg2b, NN, CC, KP, stream);
    run_gemm(hBb,   KP, Ws2b, KP, b_conv2, h2, CC, NN, CC, KP, 2, stream);
    run_gemm(agg2b, KP, Wn2b, KP, nullptr, h2, CC, NN, CC, KP, 1, stream);
    k_bn_elem<<<cdiv(nc, 256), 256, 0, stream>>>(h2, bn2_g, bn2_b, bn2_m, bn2_v, 0);

    // ---- outputs ----
    k_leaky_copy<<<cdiv(nc, 256), 256, 0, stream>>>(h2, out_x, nc);     // x_out = leaky(bn2)
    run_cast(h2, CC, 0, h2b, NN, CC, KP, stream);                       // heads use pre-leaky bn2
    run_gemm(h2b, KP, Wlinb, KP, b_lin, h3, CC, NN, CC, KP, 2, stream);
    k_relu_inplace<<<cdiv(nc, 256), 256, 0, stream>>>(h3, nc);
    run_cast(h3, CC, 0, h3b, NN, CC, KP, stream);
    run_gemm(h3b, KP, Wbceb, KP, b_bce, out_b, NBCE, NN, NBCE, KP, 2, stream);
    run_gemm(h3b, KP, Wceb,  KP, b_ce,  out_y, NCE,  NN, NCE,  KP, 2, stream);
}
